// Quetzal_18614388261681
// MI455X (gfx1250) — matrix operand order repaired and checked by reading, not compiled, not run on hardware
//
#include <hip/hip_runtime.h>
#include <hip/hip_bf16.h>
#include <math.h>

// ---------------- constants ----------------
#define SQ   2048      // S
#define DIM  768       // D
#define NH   12        // heads
#define HD   64        // head dim
#define NFQ  64        // NF
#define DMUL 4         // DIFF_MULT
#define STOPA 126
#define PADA  127

typedef __attribute__((ext_vector_type(16))) _Float16 v16h;
typedef __attribute__((ext_vector_type(8)))  float    v8f;
typedef unsigned int u32;

// ---------------- device helpers ----------------
__device__ __forceinline__ float geluf(float x) {
    const float k0 = 0.7978845608028654f; // sqrt(2/pi)
    float x3 = x * x * x;
    return 0.5f * x * (1.0f + tanhf(k0 * (x + 0.044715f * x3)));
}
__device__ __forceinline__ float siluf(float x) { return x / (1.0f + expf(-x)); }

__device__ __forceinline__ float blk_sum(float v, float* sm) {
    int t = threadIdx.x;
    sm[t] = v; __syncthreads();
    for (int s = blockDim.x >> 1; s > 0; s >>= 1) {
        if (t < s) sm[t] += sm[t + s];
        __syncthreads();
    }
    float r = sm[0]; __syncthreads();
    return r;
}
__device__ __forceinline__ float blk_max(float v, float* sm) {
    int t = threadIdx.x;
    sm[t] = v; __syncthreads();
    for (int s = blockDim.x >> 1; s > 0; s >>= 1) {
        if (t < s) sm[t] = fmaxf(sm[t], sm[t + s]);
        __syncthreads();
    }
    float r = sm[0]; __syncthreads();
    return r;
}

// ---- f16 fragment loaders (16x16x32 WMMA, contiguous k order used by this code), compile-time strides ----
// A 16x32: lanes 0-15 -> M=lane; VGPR v: K=16*hh+2v,+1
template <int LD>
__device__ __forceinline__ v16h load_a_h(const _Float16* P, int m0, int k0, int lane) {
    const int mm = m0 + (lane & 15);
    const int hh = lane >> 4;
    const u32* p = reinterpret_cast<const u32*>(P + (size_t)mm * LD + k0);
    union { v16h v; u32 u[8]; } U;
#pragma unroll
    for (int v = 0; v < 4; ++v) U.u[v] = p[8 * hh + v];
#pragma unroll
    for (int v = 4; v < 8; ++v) U.u[v] = p[8 * hh + v];
    return U.v;
}
// B 32x16 from Bt[N][K]: lanes 0-15 -> N=lane, K=0..15 (2/VGPR); lanes 16-31 -> K=16..31.
// 32 contiguous bytes/lane -> b128 loads; nofs is a compile-time tile offset so
// different n-tiles fold into immediate instruction offsets of one base address.
template <int LD, int NOFS>
__device__ __forceinline__ v16h load_b_h(const _Float16* P, int n0, int k0, int lane) {
    const int nn = n0 + (lane & 15);
    const int hh = lane >> 4;
    const u32* p = reinterpret_cast<const u32*>(P + (size_t)(nn) * LD + (size_t)NOFS * LD +
                                                k0 + 16 * hh);
    union { v16h v; u32 u[8]; } U;
#pragma unroll
    for (int v = 0; v < 8; ++v) U.u[v] = p[v];
    return U.v;
}

// ---------------- WMMA GEMM, 32x64 per wave (8 WMMA / K-step) ----------------
// C = act(scale * A[M,K](f16) x Bt[N,K](f16) + res(f32)); C f32 or f16.
// blockDim 128 (4 waves along M). Block tile 128(M) x 64(N); grid (N/64, M/128).
template <int LDA, int LDB, int ACT, bool RES, bool C16>
__global__ void q_gemm(const _Float16* __restrict__ A,
                       const _Float16* __restrict__ B,
                       void* __restrict__ Cp, int ldc,
                       const float* __restrict__ res, int ldr,
                       int K, float scale) {
    const int lane = threadIdx.x & 31;
    const int wave = threadIdx.x >> 5;
    const int m0 = (blockIdx.y * 4 + wave) * 32;
    const int n0 = blockIdx.x * 64;

    v8f acc[2][4];
#pragma unroll
    for (int i = 0; i < 2; ++i)
#pragma unroll
        for (int j = 0; j < 4; ++j) acc[i][j] = (v8f){};

    for (int k0 = 0; k0 < K; k0 += 32) {
        if (k0 + 32 < K) {
            const int mm = m0 + (lane & 15);
            __builtin_prefetch(A + (size_t)mm * LDA + k0 + 32 + (lane >> 4) * 16, 0, 1);
        }
        v16h a0 = load_a_h<LDA>(A, m0,      k0, lane);
        v16h a1 = load_a_h<LDA>(A, m0 + 16, k0, lane);
        v16h b0 = load_b_h<LDB, 0 >(B, n0, k0, lane);
        v16h b1 = load_b_h<LDB, 16>(B, n0, k0, lane);
        v16h b2 = load_b_h<LDB, 32>(B, n0, k0, lane);
        v16h b3 = load_b_h<LDB, 48>(B, n0, k0, lane);
        acc[0][0] = __builtin_amdgcn_wmma_f32_16x16x32_f16(false, a0, false, b0, (short)0, acc[0][0], false, false);
        acc[0][1] = __builtin_amdgcn_wmma_f32_16x16x32_f16(false, a0, false, b1, (short)0, acc[0][1], false, false);
        acc[0][2] = __builtin_amdgcn_wmma_f32_16x16x32_f16(false, a0, false, b2, (short)0, acc[0][2], false, false);
        acc[0][3] = __builtin_amdgcn_wmma_f32_16x16x32_f16(false, a0, false, b3, (short)0, acc[0][3], false, false);
        acc[1][0] = __builtin_amdgcn_wmma_f32_16x16x32_f16(false, a1, false, b0, (short)0, acc[1][0], false, false);
        acc[1][1] = __builtin_amdgcn_wmma_f32_16x16x32_f16(false, a1, false, b1, (short)0, acc[1][1], false, false);
        acc[1][2] = __builtin_amdgcn_wmma_f32_16x16x32_f16(false, a1, false, b2, (short)0, acc[1][2], false, false);
        acc[1][3] = __builtin_amdgcn_wmma_f32_16x16x32_f16(false, a1, false, b3, (short)0, acc[1][3], false, false);
    }

    const int hh = lane >> 4;
    const int nl = lane & 15;
#pragma unroll
    for (int i = 0; i < 2; ++i) {
#pragma unroll
        for (int j = 0; j < 4; ++j) {
            const int nn = n0 + 16 * j + nl;
#pragma unroll
            for (int v = 0; v < 8; ++v) {
                int mm = m0 + 16 * i + v + 8 * hh;
                float val = acc[i][j][v] * scale;
                if (RES) val += res[(size_t)mm * (size_t)ldr + nn];
                if (ACT == 1) val = geluf(val);
                else if (ACT == 2) val = siluf(val);
                if (C16) ((_Float16*)Cp)[(size_t)mm * (size_t)ldc + nn] = (_Float16)val;
                else     ((float*)Cp)[(size_t)mm * (size_t)ldc + nn] = val;
            }
        }
    }
}

// 16x16 per wave variant (narrow-N P.V GEMM: parallelism along M).
// blockDim 128 (4 waves along N); grid (N/64, M/16).
template <int LDA, int LDB, bool C16>
__global__ void q_gemm16(const _Float16* __restrict__ A,
                         const _Float16* __restrict__ B,
                         void* __restrict__ Cp, int ldc, int K, float scale) {
    const int lane = threadIdx.x & 31;
    const int wave = threadIdx.x >> 5;
    const int n0 = (blockIdx.x * 4 + wave) * 16;
    const int m0 = blockIdx.y * 16;
    v8f acc = {};
    for (int k0 = 0; k0 < K; k0 += 32) {
        if (k0 + 32 < K) {
            const int mm = m0 + (lane & 15);
            __builtin_prefetch(A + (size_t)mm * LDA + k0 + 32 + (lane >> 4) * 16, 0, 1);
        }
        v16h a = load_a_h<LDA>(A, m0, k0, lane);
        v16h b = load_b_h<LDB, 0>(B, n0, k0, lane);
        acc = __builtin_amdgcn_wmma_f32_16x16x32_f16(false, a, false, b, (short)0, acc, false, false);
    }
    const int hh = lane >> 4;
    const int nn = n0 + (lane & 15);
#pragma unroll
    for (int v = 0; v < 8; ++v) {
        int mm = m0 + v + 8 * hh;
        float val = acc[v] * scale;
        if (C16) ((_Float16*)Cp)[(size_t)mm * (size_t)ldc + nn] = (_Float16)val;
        else     ((float*)Cp)[(size_t)mm * (size_t)ldc + nn] = val;
    }
}

// ---------------- conversion kernels ----------------
// W f32 [K][ldw] column slice (colofs..colofs+N) -> Wt f16 [N][K]
__global__ void q_w2h(const float* __restrict__ W, int ldw, int colofs,
                      _Float16* __restrict__ Wt, int N, int K) {
    int i = blockIdx.x * 256 + threadIdx.x;
    if (i >= N * K) return;
    int n = i / K, k = i - n * K;
    Wt[i] = (_Float16)W[(size_t)k * (size_t)ldw + colofs + n];
}
__global__ void q_f2h(const float* __restrict__ in, _Float16* __restrict__ out, int n) {
    for (int i = blockIdx.x * 256 + threadIdx.x; i < n; i += gridDim.x * 256)
        out[i] = (_Float16)in[i];
}
// VT[d][s] = QKVh[s][2D + d]  (f16 transpose of the V section)
__global__ void q_vt(const _Float16* __restrict__ QKVh, _Float16* __restrict__ VT) {
    int i = blockIdx.x * 256 + threadIdx.x;
    if (i >= SQ * DIM) return;
    int d = i / SQ, s = i - d * SQ;
    VT[i] = QKVh[(size_t)s * (3 * DIM) + 2 * DIM + d];
}

// ---------------- small kernels ----------------
__global__ void q_zero_out(float* out) { out[0] = 0.0f; out[1] = 0.0f; }

__global__ void q_doc(const int* __restrict__ atoms, int* doc, int* pad) {
    int c = 0;
    for (int i = 0; i < SQ; ++i) {
        doc[i] = c;
        if (atoms[i] == STOPA) c++;
        pad[i] = (atoms[i] == PADA) ? 1 : 0;
    }
}

__global__ void q_embed(const float* __restrict__ wpe, const float* __restrict__ e_atoms,
                        const float* __restrict__ w_coords, const float* __restrict__ four_B,
                        const float* __restrict__ four_W, const float* __restrict__ mp_phases,
                        const int* __restrict__ idx, const int* __restrict__ atoms,
                        const float* __restrict__ coords, float* __restrict__ seq) {
    __shared__ float feat[2 * NFQ];
    const int s = blockIdx.x;
    const float c0 = coords[s * 3 + 0], c1 = coords[s * 3 + 1], c2 = coords[s * 3 + 2];
    const float TWO_PI = 6.283185307179586f;
    for (int f = threadIdx.x; f < NFQ; f += blockDim.x) {
        float pr = TWO_PI * (c0 * four_B[f] + c1 * four_B[NFQ + f] + c2 * four_B[2 * NFQ + f]);
        feat[f]       = cosf(pr);
        feat[NFQ + f] = sinf(pr);
    }
    __syncthreads();
    const int a = atoms[s];
    const int pe = idx[s];
    const float SQ2 = 1.4142135623730951f;
    for (int d = threadIdx.x; d < DIM; d += blockDim.x) {
        float v = e_atoms[(size_t)a * DIM + d]
                + c0 * w_coords[d] + c1 * w_coords[DIM + d] + c2 * w_coords[2 * DIM + d]
                + SQ2 * cosf(mp_phases[d])
                + wpe[(size_t)pe * DIM + d];
#pragma unroll 4
        for (int f = 0; f < 2 * NFQ; ++f) v += feat[f] * four_W[(size_t)f * DIM + d];
        seq[(size_t)s * DIM + d] = v;
    }
}

// LayerNorm: f32 in -> f16 out (feeds GEMM A side)
__global__ void q_ln(const float* __restrict__ x, const float* __restrict__ g,
                     const float* __restrict__ b, _Float16* __restrict__ out) {
    __shared__ float sm[256];
    const int r = blockIdx.x;
    const float* row = x + (size_t)r * DIM;
    float s = 0.0f, s2 = 0.0f;
    for (int d = threadIdx.x; d < DIM; d += 256) { float v = row[d]; s += v; s2 += v * v; }
    float mean = blk_sum(s, sm) * (1.0f / DIM);
    float msq  = blk_sum(s2, sm) * (1.0f / DIM);
    float inv = rsqrtf(msq - mean * mean + 1e-5f);
    for (int d = threadIdx.x; d < DIM; d += 256) {
        float v = (row[d] - mean) * inv;
        out[(size_t)r * DIM + d] = (_Float16)(v * g[d] + b[d]);
    }
}

// masked softmax: f32 scores in, f16 probs out
__global__ void q_softmax(const float* __restrict__ sc, _Float16* __restrict__ ph,
                          const int* __restrict__ doc, const int* __restrict__ pad) {
    __shared__ float sm[256];
    const int q = blockIdx.x;
    const float* row = sc + (size_t)q * SQ;
    _Float16* orow = ph + (size_t)q * SQ;
    const int dq = doc[q];
    const int pq = pad[q];
    float mx = -3.0e38f;
    for (int j = threadIdx.x; j < SQ; j += 256) {
        bool valid = (j <= q) && (doc[j] == dq) && (!pq) && (!pad[j]);
        if (valid) mx = fmaxf(mx, row[j]);
    }
    mx = blk_max(mx, sm);
    float sum = 0.0f;
    for (int j = threadIdx.x; j < SQ; j += 256) {
        bool valid = (j <= q) && (doc[j] == dq) && (!pq) && (!pad[j]);
        if (valid) sum += expf(row[j] - mx);
    }
    sum = blk_sum(sum, sm);
    float inv = (sum > 0.0f) ? (1.0f / sum) : 0.0f;
    for (int j = threadIdx.x; j < SQ; j += 256) {
        bool valid = (j <= q) && (doc[j] == dq) && (!pq) && (!pad[j]);
        orow[j] = (_Float16)(valid ? expf(row[j] - mx) * inv : 0.0f);
    }
}

__global__ void q_ce(const float* __restrict__ logits, const int* __restrict__ tgt,
                     const float* __restrict__ mask, float* __restrict__ out) {
    __shared__ float sm[128];
    const int s = blockIdx.x;
    float v = logits[(size_t)s * 128 + threadIdx.x];
    float mx = blk_max(v, sm);
    float se = blk_sum(expf(v - mx), sm);
    if (threadIdx.x == 0) {
        float lse = mx + logf(se);
        float ce = lse - logits[(size_t)s * 128 + tgt[s]];
        atomicAdd(out + 1, ce * mask[s] * (1.0f / SQ));
    }
}

__global__ void q_addemb(const float* __restrict__ seq, const float* __restrict__ e_atoms,
                         const int* __restrict__ tgt, float* __restrict__ z) {
    const int s = blockIdx.x;
    const int a = tgt[s];
    for (int d = threadIdx.x; d < DIM; d += 256)
        z[(size_t)s * DIM + d] = seq[(size_t)s * DIM + d] + e_atoms[(size_t)a * DIM + d];
}

// ---- diffusion MLP helpers ----
__global__ void q_prep(const float* __restrict__ rnd, const float* __restrict__ noise,
                       const float* __restrict__ tc, float* wgt, float* cskip, float* cout_,
                       float* cnoise, float* xin, float* noisy) {
    int r = blockIdx.x * blockDim.x + threadIdx.x;
    if (r >= SQ) return;
    float sg = expf(rnd[r] * 1.2f - 1.2f);
    float s2 = sg * sg;
    wgt[r]   = (s2 + 1.0f) / s2;
    cskip[r] = 1.0f / (s2 + 1.0f);
    cout_[r] = sg * rsqrtf(s2 + 1.0f);
    float cin = rsqrtf(1.0f + s2);
    cnoise[r] = logf(sg) * 0.25f;
#pragma unroll
    for (int j = 0; j < 3; ++j) {
        float ny = tc[r * 3 + j] + noise[r * 3 + j] * sg;
        noisy[r * 3 + j] = ny;
        xin[r * 3 + j]   = cin * ny;
    }
}

__global__ void q_hin(const float* __restrict__ xin, const float* __restrict__ w_in,
                      float* __restrict__ h) {
    const int r = blockIdx.x;
    float a = xin[r * 3], b = xin[r * 3 + 1], c = xin[r * 3 + 2];
    for (int d = threadIdx.x; d < DIM; d += 256)
        h[(size_t)r * DIM + d] = a * w_in[d] + b * w_in[DIM + d] + c * w_in[2 * DIM + d];
}

__global__ void q_temb(const float* __restrict__ cnoise, const float* __restrict__ t1,
                       _Float16* __restrict__ out) {
    const int r = blockIdx.x;
    float t = cnoise[r];
    for (int d = threadIdx.x; d < DIM; d += 256)
        out[(size_t)r * DIM + d] = (_Float16)siluf(t * t1[d]);
}

__global__ void q_silu_ew(const float* __restrict__ in, _Float16* __restrict__ out, int n) {
    for (int i = blockIdx.x * blockDim.x + threadIdx.x; i < n; i += gridDim.x * blockDim.x)
        out[i] = (_Float16)siluf(in[i]);
}

__global__ void q_gateadd(float* __restrict__ h, const float* __restrict__ ga,
                          const float* __restrict__ t, int n) {
    for (int i = blockIdx.x * blockDim.x + threadIdx.x; i < n; i += gridDim.x * blockDim.x)
        h[i] += ga[i] * t[i];
}

// AdaLN: out_h (f16, feeds GEMM) and out_f (f32, feeds loss)
__global__ void q_adaln(const float* __restrict__ h, const float* __restrict__ sh,
                        const float* __restrict__ sc, _Float16* __restrict__ out_h,
                        float* __restrict__ out_f) {
    __shared__ float sm[256];
    const int r = blockIdx.x;
    const float* row = h + (size_t)r * DIM;
    float s = 0.0f, s2 = 0.0f;
    for (int d = threadIdx.x; d < DIM; d += 256) { float v = row[d]; s += v; s2 += v * v; }
    float mean = blk_sum(s, sm) * (1.0f / DIM);
    float msq  = blk_sum(s2, sm) * (1.0f / DIM);
    float inv = rsqrtf(msq - mean * mean + 1e-5f);
    for (int d = threadIdx.x; d < DIM; d += 256) {
        float v = (row[d] - mean) * inv;
        size_t o = (size_t)r * DIM + d;
        float y = v * (1.0f + sc[o]) + sh[o];
        out_h[o] = (_Float16)y;
        out_f[o] = y;
    }
}

__global__ void q_loss(const float* __restrict__ h2, const float* __restrict__ w_out,
                       const float* __restrict__ noisy, const float* __restrict__ tc,
                       const float* __restrict__ cskip, const float* __restrict__ cout_,
                       const float* __restrict__ wgt, const float* __restrict__ mask,
                       float* __restrict__ out) {
    __shared__ float sm[256];
    const int r = blockIdx.x;
    float f0 = 0.0f, f1 = 0.0f, f2 = 0.0f;
    for (int d = threadIdx.x; d < DIM; d += 256) {
        float v = h2[(size_t)r * DIM + d];
        f0 += v * w_out[d * 3 + 0];
        f1 += v * w_out[d * 3 + 1];
        f2 += v * w_out[d * 3 + 2];
    }
    f0 = blk_sum(f0, sm); f1 = blk_sum(f1, sm); f2 = blk_sum(f2, sm);
    if (threadIdx.x == 0) {
        float F[3] = {f0, f1, f2};
        float acc = 0.0f;
        for (int j = 0; j < 3; ++j) {
            float Dx = cskip[r] * noisy[r * 3 + j] + cout_[r] * F[j];
            float dd = Dx - tc[r * 3 + j];
            acc += wgt[r] * dd * dd;
        }
        atomicAdd(out + 0, acc * mask[r] * (1.0f / (DMUL * SQ * 3)));
    }
}

// ---------------- host-side helpers ----------------
template <int LDA, int LDB, int ACT, bool RES, bool C16>
static void launch_gemm(hipStream_t st, const _Float16* A, const _Float16* B,
                        void* C, int ldc, const float* res, int ldr,
                        int M, int N, int K, float scale) {
    dim3 g(N / 64, M / 128);
    q_gemm<LDA, LDB, ACT, RES, C16><<<g, 128, 0, st>>>(A, B, C, ldc, res, ldr, K, scale);
}
static void launch_w2h(hipStream_t st, const float* W, int ldw, int colofs,
                       _Float16* Wt, int N, int K) {
    q_w2h<<<(N * K + 255) / 256, 256, 0, st>>>(W, ldw, colofs, Wt, N, K);
}

#define D3C (3 * DIM)
#define D4C (4 * DIM)

static void run_attn(hipStream_t st, float* x, const float* g1, const float* b1,
                     const float* wqkv, const float* wo, const float* g2, const float* b2,
                     const float* wfc, const float* wpr,
                     _Float16* HBh, _Float16* QKVh, float* SC, _Float16* PH, _Float16* VTh,
                     _Float16* OBh, _Float16* FCBh, _Float16* WT,
                     const int* doc, const int* pad) {
    q_ln<<<SQ, 256, 0, st>>>(x, g1, b1, HBh);
    launch_w2h(st, wqkv, D3C, 0, WT, D3C, DIM);
    launch_gemm<DIM, DIM, 0, false, true>(st, HBh, WT, QKVh, D3C, nullptr, 0, SQ, D3C, DIM, 1.0f);
    q_vt<<<(SQ * DIM + 255) / 256, 256, 0, st>>>(QKVh, VTh);
    for (int h = 0; h < NH; ++h) {
        // scores = (Q K^T)/sqrt(hd): A = Q rows, "Bt" = K rows (both f16 in QKVh)
        launch_gemm<D3C, D3C, 0, false, false>(st, QKVh + h * HD, QKVh + DIM + h * HD,
                                               SC, SQ, nullptr, 0, SQ, SQ, HD, 0.125f);
        q_softmax<<<SQ, 256, 0, st>>>(SC, PH, doc, pad);
        // O_h = P V : Bt = V^T rows (dim-major)
        dim3 gav(HD / 64, SQ / 16);
        q_gemm16<SQ, SQ, true><<<gav, 128, 0, st>>>(PH, VTh + (size_t)h * HD * SQ,
                                                    OBh + h * HD, DIM, SQ, 1.0f);
    }
    launch_w2h(st, wo, DIM, 0, WT, DIM, DIM);
    launch_gemm<DIM, DIM, 0, true, false>(st, OBh, WT, x, DIM, x, DIM, SQ, DIM, DIM, 1.0f);
    q_ln<<<SQ, 256, 0, st>>>(x, g2, b2, HBh);
    launch_w2h(st, wfc, D4C, 0, WT, D4C, DIM);
    launch_gemm<DIM, DIM, 1, false, true>(st, HBh, WT, FCBh, D4C, nullptr, 0, SQ, D4C, DIM, 1.0f);
    launch_w2h(st, wpr, DIM, 0, WT, DIM, D4C);
    launch_gemm<D4C, D4C, 0, true, false>(st, FCBh, WT, x, DIM, x, DIM, SQ, DIM, D4C, 1.0f);
}

extern "C" void kernel_launch(void* const* d_in, const int* in_sizes, int n_in,
                              void* d_out, int out_size, void* d_ws, size_t ws_size,
                              hipStream_t stream) {
    // ---- input pointers (setup_inputs depth-first dict order) ----
    const float* wpe       = (const float*)d_in[0];
    const float* e_atoms   = (const float*)d_in[1];
    const float* w_coords  = (const float*)d_in[2];
    const float* four_B    = (const float*)d_in[3];
    const float* four_W    = (const float*)d_in[4];
    // d_in[5] = mp_freqs (unused: MPFourier of zeros)
    const float* mp_phases = (const float*)d_in[6];
    const float* w_logits  = (const float*)d_in[7];
    const float* mlp_w_in  = (const float*)d_in[40];
    const float* mlp_t1    = (const float*)d_in[41];
    const float* mlp_t2    = (const float*)d_in[42];
    const float* mlp_cond  = (const float*)d_in[43];
    const float* mlp_ada[2]  = {(const float*)d_in[44], (const float*)d_in[47]};
    const float* mlp_fc1[2]  = {(const float*)d_in[45], (const float*)d_in[48]};
    const float* mlp_fc2[2]  = {(const float*)d_in[46], (const float*)d_in[49]};
    const float* mlp_adaf  = (const float*)d_in[50];
    const float* mlp_wout  = (const float*)d_in[51];
    const int*   idx       = (const int*)d_in[52];
    const int*   atoms     = (const int*)d_in[53];
    const float* coords    = (const float*)d_in[54];
    const int*   tgt_atoms = (const int*)d_in[55];
    const float* tgt_crds  = (const float*)d_in[56];
    const float* loss_mask = (const float*)d_in[57];
    const float* rnd_norm  = (const float*)d_in[58];
    const float* noise     = (const float*)d_in[59];
    float* out = (float*)d_out;

    // ---- workspace layout ----
    const size_t SD = (size_t)SQ * DIM;
    float* ws = (float*)d_ws;
    size_t o = 0;
    float* SEQ = ws + o;  o += SD;
    float* Z   = ws + o;  o += SD;
    float* SC  = ws + o;  o += (size_t)SQ * SQ;
    float* LG  = ws + o;  o += (size_t)SQ * 128;
    float* HR  = ws + o;  o += SD;
    float* YR  = ws + o;  o += SD;
    float* SHR = ws + o;  o += SD;
    float* SCR = ws + o;  o += SD;
    float* GAR = ws + o;  o += SD;
    float* H2R = ws + o;  o += SD;
    int*   doc = (int*)(ws + o); o += SQ;
    int*   pad = (int*)(ws + o); o += SQ;
    float* WGT = ws + o;  o += SQ;
    float* CSK = ws + o;  o += SQ;
    float* COU = ws + o;  o += SQ;
    float* CNS = ws + o;  o += SQ;
    float* XIN = ws + o;  o += 3 * SQ;
    float* NSY = ws + o;  o += 3 * SQ;
    // f16 region (2 halfs per float slot)
    _Float16* hbase = (_Float16*)(ws + o);
    size_t ho = 0;
    _Float16* HBh  = hbase + ho; ho += SD;            // ln out / seq f16 scratch
    _Float16* QKVh = hbase + ho; ho += 3 * SD;
    _Float16* PH   = hbase + ho; ho += (size_t)SQ * SQ;
    _Float16* VTh  = hbase + ho; ho += SD;
    _Float16* OBh  = hbase + ho; ho += SD;
    _Float16* FCBh = hbase + ho; ho += 4 * SD;
    _Float16* ZH   = hbase + ho; ho += SD;
    _Float16* SYRh = hbase + ho; ho += SD;
    _Float16* H2Rh = hbase + ho; ho += SD;
    _Float16* FC1h = hbase + ho; ho += SD;
    _Float16* WT   = hbase + ho; ho += 4 * SD;        // per-use weight transpose scratch
    _Float16* T2H  = hbase + ho; ho += (size_t)DIM * DIM;
    _Float16* CDH  = hbase + ho; ho += (size_t)DIM * DIM;
    _Float16* ADAH[6];
    for (int i = 0; i < 6; ++i) { ADAH[i] = hbase + ho; ho += (size_t)DIM * DIM; }
    _Float16* FC1H[2]; _Float16* FC2H[2];
    for (int i = 0; i < 2; ++i) { FC1H[i] = hbase + ho; ho += (size_t)DIM * DIM; }
    for (int i = 0; i < 2; ++i) { FC2H[i] = hbase + ho; ho += (size_t)DIM * DIM; }
    _Float16* ADFH[2];
    for (int i = 0; i < 2; ++i) { ADFH[i] = hbase + ho; ho += (size_t)DIM * DIM; }
    (void)ws_size; (void)in_sizes; (void)n_in; (void)out_size;

    q_zero_out<<<1, 1, 0, stream>>>(out);
    q_doc<<<1, 1, 0, stream>>>(atoms, doc, pad);
    q_embed<<<SQ, 256, 0, stream>>>(wpe, e_atoms, w_coords, four_B, four_W, mp_phases,
                                    idx, atoms, coords, SEQ);

    // encode1 blocks
    for (int L = 0; L < 2; ++L) {
        int b = 8 + 8 * L;
        run_attn(stream, SEQ,
                 (const float*)d_in[b + 0], (const float*)d_in[b + 1],
                 (const float*)d_in[b + 2], (const float*)d_in[b + 3],
                 (const float*)d_in[b + 4], (const float*)d_in[b + 5],
                 (const float*)d_in[b + 6], (const float*)d_in[b + 7],
                 HBh, QKVh, SC, PH, VTh, OBh, FCBh, WT, doc, pad);
    }

    // logits + CE  (A = f16 copy of seq)
    q_f2h<<<(int)((SD + 255) / 256), 256, 0, stream>>>(SEQ, HBh, (int)SD);
    launch_w2h(stream, w_logits, 128, 0, WT, 128, DIM);
    launch_gemm<DIM, DIM, 0, false, false>(stream, HBh, WT, LG, 128, nullptr, 0,
                                           SQ, 128, DIM, 1.0f);
    q_ce<<<SQ, 128, 0, stream>>>(LG, tgt_atoms, loss_mask, out);

    // encode2
    q_addemb<<<SQ, 256, 0, stream>>>(SEQ, e_atoms, tgt_atoms, Z);
    for (int L = 0; L < 2; ++L) {
        int b = 24 + 8 * L;
        run_attn(stream, Z,
                 (const float*)d_in[b + 0], (const float*)d_in[b + 1],
                 (const float*)d_in[b + 2], (const float*)d_in[b + 3],
                 (const float*)d_in[b + 4], (const float*)d_in[b + 5],
                 (const float*)d_in[b + 6], (const float*)d_in[b + 7],
                 HBh, QKVh, SC, PH, VTh, OBh, FCBh, WT, doc, pad);
    }

    // ---- diffusion MLP ----
    const int NEL = SQ * DIM;
    const int GEW = (NEL + 255) / 256;
    // pre-convert MLP weights once (reused over DMUL replicas)
    q_f2h<<<GEW, 256, 0, stream>>>(Z, ZH, NEL);
    launch_w2h(stream, mlp_t2, DIM, 0, T2H, DIM, DIM);
    launch_w2h(stream, mlp_cond, DIM, 0, CDH, DIM, DIM);
    for (int rb = 0; rb < 2; ++rb) {
        for (int j = 0; j < 3; ++j)
            launch_w2h(stream, mlp_ada[rb], D3C, j * DIM, ADAH[rb * 3 + j], DIM, DIM);
        launch_w2h(stream, mlp_fc1[rb], DIM, 0, FC1H[rb], DIM, DIM);
        launch_w2h(stream, mlp_fc2[rb], DIM, 0, FC2H[rb], DIM, DIM);
    }
    for (int j = 0; j < 2; ++j)
        launch_w2h(stream, mlp_adaf, 2 * DIM, j * DIM, ADFH[j], DIM, DIM);

    for (int dm = 0; dm < DMUL; ++dm) {
        q_prep<<<(SQ + 255) / 256, 256, 0, stream>>>(rnd_norm + dm * SQ,
                                                     noise + (size_t)dm * SQ * 3,
                                                     tgt_crds, WGT, CSK, COU, CNS, XIN, NSY);
        q_hin<<<SQ, 256, 0, stream>>>(XIN, mlp_w_in, HR);
        q_temb<<<SQ, 256, 0, stream>>>(CNS, mlp_t1, SYRh);
        launch_gemm<DIM, DIM, 0, false, false>(stream, SYRh, T2H, YR, DIM, nullptr, 0,
                                               SQ, DIM, DIM, 1.0f);
        launch_gemm<DIM, DIM, 0, true, false>(stream, ZH, CDH, YR, DIM, YR, DIM,
                                              SQ, DIM, DIM, 1.0f);
        for (int rb = 0; rb < 2; ++rb) {
            q_silu_ew<<<GEW, 256, 0, stream>>>(YR, SYRh, NEL);
            launch_gemm<DIM, DIM, 0, false, false>(stream, SYRh, ADAH[rb * 3 + 0], SHR, DIM,
                                                   nullptr, 0, SQ, DIM, DIM, 1.0f);
            launch_gemm<DIM, DIM, 0, false, false>(stream, SYRh, ADAH[rb * 3 + 1], SCR, DIM,
                                                   nullptr, 0, SQ, DIM, DIM, 1.0f);
            launch_gemm<DIM, DIM, 0, false, false>(stream, SYRh, ADAH[rb * 3 + 2], GAR, DIM,
                                                   nullptr, 0, SQ, DIM, DIM, 1.0f);
            q_adaln<<<SQ, 256, 0, stream>>>(HR, SHR, SCR, H2Rh, H2R);
            launch_gemm<DIM, DIM, 2, false, true>(stream, H2Rh, FC1H[rb], FC1h, DIM,
                                                  nullptr, 0, SQ, DIM, DIM, 1.0f);
            launch_gemm<DIM, DIM, 0, false, false>(stream, FC1h, FC2H[rb], SCR, DIM,
                                                   nullptr, 0, SQ, DIM, DIM, 1.0f);
            q_gateadd<<<GEW, 256, 0, stream>>>(HR, GAR, SCR, NEL);
        }
        q_silu_ew<<<GEW, 256, 0, stream>>>(YR, SYRh, NEL);
        launch_gemm<DIM, DIM, 0, false, false>(stream, SYRh, ADFH[0], SHR, DIM,
                                               nullptr, 0, SQ, DIM, DIM, 1.0f);
        launch_gemm<DIM, DIM, 0, false, false>(stream, SYRh, ADFH[1], SCR, DIM,
                                               nullptr, 0, SQ, DIM, DIM, 1.0f);
        q_adaln<<<SQ, 256, 0, stream>>>(HR, SHR, SCR, H2Rh, H2R);
        q_loss<<<SQ, 256, 0, stream>>>(H2R, mlp_wout, NSY, tgt_crds, CSK, COU, WGT,
                                       loss_mask, out);
    }
}
